// RNN_23132693856284
// MI455X (gfx1250) — matrix operand order repaired and checked by reading, not compiled, not run on hardware
//
#include <hip/hip_runtime.h>
#include <hip/hip_bf16.h>

// ---------------------------------------------------------------------------
// Problem constants
// ---------------------------------------------------------------------------
#define BATCH     512
#define TSTEPS    128
#define SSIZE     512      // STATE_SIZE
#define HSIZE     1536     // RNN_SIZE
#define KTOT      2048     // HSIZE + SSIZE (combined K for the recurrent GEMM)
#define KT_H      48       // HSIZE / 32
#define KT_X      16       // SSIZE / 32
#define KT_ALL    64       // KTOT / 32
#define NT_H      96       // HSIZE / 16
#define NT_S      32       // SSIZE / 16
#define MT_B      32       // BATCH / 16

typedef float v8f  __attribute__((ext_vector_type(8)));
typedef __bf16 v16bf __attribute__((ext_vector_type(16)));
typedef int v4i_ __attribute__((vector_size(16)));   // matches async builtin params

union AB16 { uint4 q[2]; v16bf v; };
union PK16 { unsigned short s[16]; uint4 q[2]; };

#if __has_builtin(__builtin_amdgcn_global_load_async_to_lds_b128) && \
    __has_builtin(__builtin_amdgcn_s_wait_asynccnt)
#define USE_ASYNC_LDS 1
#define GASYNC_B128(gp, lp) \
  __builtin_amdgcn_global_load_async_to_lds_b128((v4i_*)(gp), (v4i_*)(lp), 0, 0)
#else
#define USE_ASYNC_LDS 0
#endif

// round-to-nearest-even fp32 -> bf16 bits
static __device__ __forceinline__ unsigned short f2bf(float f) {
  unsigned int u = __float_as_uint(f);
  u += 0x7fffu + ((u >> 16) & 1u);
  return (unsigned short)(u >> 16);
}

// A-matrix (16x32 bf16) per-lane element j -> K offset within the 32-wide tile
static __device__ __forceinline__ int kpatA(int lane, int j) {
  int khalf = lane >> 4;
  return (j < 8) ? (khalf * 8 + j) : (16 + khalf * 8 + (j - 8));
}
// B-matrix (32x16 bf16): same K pattern as A (lanes 0-15 hold K 0..7,16..23; lanes 16-31 hold K 8..15,24..31)
static __device__ __forceinline__ int kpatB(int lane, int j) {
  int khalf = lane >> 4;
  return (j < 8) ? (khalf * 8 + j) : (16 + khalf * 8 + (j - 8));
}

static __device__ __forceinline__ v16bf load16bf(const unsigned short* p) {
  AB16 a;
  a.q[0] = ((const uint4*)p)[0];
  a.q[1] = ((const uint4*)p)[1];
  return a.v;
}
static __device__ __forceinline__ v16bf load16bf_lds(const unsigned short* p) {
  AB16 a;
  a.q[0] = ((const uint4*)p)[0];
  a.q[1] = ((const uint4*)p)[1];
  return a.v;
}

// ---------------------------------------------------------------------------
// Zero accumulators + zero h packed buffer 0 (initial hidden state = 0)
// ---------------------------------------------------------------------------
__global__ __launch_bounds__(256) void zero_kernel(unsigned int* hz, int n, float* accum) {
  int tid = blockIdx.x * blockDim.x + threadIdx.x;
  if (tid < 16) accum[tid] = 0.f;
  for (int i = tid; i < n; i += gridDim.x * blockDim.x) hz[i] = 0u;
}

// ---------------------------------------------------------------------------
// Pack W = [W_h ; W_in] (2048 x 1536, fp32 row-major) into bf16 B-layout tiles
// ---------------------------------------------------------------------------
__global__ __launch_bounds__(256) void packW_kernel(const float* __restrict__ Wh,
                                                    const float* __restrict__ Win,
                                                    unsigned short* __restrict__ Wp) {
  int wv = (blockIdx.x * blockDim.x + threadIdx.x) >> 5;   // 6144 tiles
  int lane = threadIdx.x & 31;
  if (wv >= NT_H * KT_ALL) return;
  int nt = wv / KT_ALL, kt = wv % KT_ALL;
  int n = nt * 16 + (lane & 15);
  PK16 p;
#pragma unroll
  for (int j = 0; j < 16; ++j) {
    int k = kt * 32 + kpatB(lane, j);
    float v = (k < HSIZE) ? Wh[(size_t)k * HSIZE + n]
                          : Win[(size_t)(k - HSIZE) * HSIZE + n];
    p.s[j] = f2bf(v);
  }
  uint4* dst = (uint4*)(Wp + (((size_t)(nt * KT_ALL + kt)) << 9) + lane * 16);
  dst[0] = p.q[0]; dst[1] = p.q[1];
}

// ---------------------------------------------------------------------------
// Pack W_out (1536 x 512) into bf16 B-layout tiles
// ---------------------------------------------------------------------------
__global__ __launch_bounds__(256) void packWout_kernel(const float* __restrict__ Wout,
                                                       unsigned short* __restrict__ WoutP) {
  int wv = (blockIdx.x * blockDim.x + threadIdx.x) >> 5;   // 1536 tiles
  int lane = threadIdx.x & 31;
  if (wv >= NT_S * KT_H) return;
  int nt = wv / KT_H, kt = wv % KT_H;
  int n = nt * 16 + (lane & 15);
  PK16 p;
#pragma unroll
  for (int j = 0; j < 16; ++j) {
    int k = kt * 32 + kpatB(lane, j);
    p.s[j] = f2bf(Wout[(size_t)k * SSIZE + n]);
  }
  uint4* dst = (uint4*)(WoutP + (((size_t)(nt * KT_H + kt)) << 9) + lane * 16);
  dst[0] = p.q[0]; dst[1] = p.q[1];
}

// ---------------------------------------------------------------------------
// Weight L1 loss partial sums
// ---------------------------------------------------------------------------
__global__ __launch_bounds__(256) void wloss_kernel(const float* __restrict__ Wh,
                                                    float* __restrict__ accum) {
  int tid = blockIdx.x * blockDim.x + threadIdx.x;
  int stride = gridDim.x * blockDim.x;
  float sab = 0.f, sba = 0.f, sbb = 0.f;
  for (int idx = tid; idx < HSIZE * HSIZE; idx += stride) {
    int r = idx / HSIZE, c = idx % HSIZE;
    float a = fabsf(Wh[idx]);
    if (c >= SSIZE) sab += a;
    if (r >= SSIZE) sba += a;
    if (r >= SSIZE && c >= SSIZE) sbb += a;
  }
#pragma unroll
  for (int off = 16; off; off >>= 1) {
    sab += __shfl_xor(sab, off, 32);
    sba += __shfl_xor(sba, off, 32);
    sbb += __shfl_xor(sbb, off, 32);
  }
  if ((threadIdx.x & 31) == 0) {
    atomicAdd(&accum[2], sab);
    atomicAdd(&accum[3], sba);
    atomicAdd(&accum[4], sbb);
  }
}

// ---------------------------------------------------------------------------
// Pack x[:, t, :] (512 x 512 fp32) into bf16 A-layout tiles
// ---------------------------------------------------------------------------
__global__ __launch_bounds__(256) void packx_kernel(const float* __restrict__ x,
                                                    unsigned short* __restrict__ xA,
                                                    int t) {
  int wv = (blockIdx.x * blockDim.x + threadIdx.x) >> 5;   // 512 tiles
  int lane = threadIdx.x & 31;
  if (wv >= MT_B * KT_X) return;
  int mt = wv >> 4, ktx = wv & 15;
  int m = mt * 16 + (lane & 15);
  const float* src = x + ((size_t)m * TSTEPS + t) * SSIZE;
  PK16 p;
#pragma unroll
  for (int j = 0; j < 16; ++j) {
    int k = ktx * 32 + kpatA(lane, j);
    p.s[j] = f2bf(src[k]);
  }
  uint4* dst = (uint4*)(xA + (((size_t)(mt * KT_X + ktx)) << 9) + lane * 16);
  dst[0] = p.q[0]; dst[1] = p.q[1];
}

// ---------------------------------------------------------------------------
// Recurrent step GEMM: h_next = relu([h, x_t] @ [W_h;W_in] + b)
// Grid (24, 8), 128 threads (4 waves). Tile: 64 rows x 64 cols.
// B staged via async-copy into an 8-deep LDS ring (issue distance 3).
// B fragments are register-double-buffered across K-steps: after each
// barrier, ds-load chunk kt+1's fragments while WMMAs consume chunk kt's
// fragments from registers -> no dscnt stall in the WMMA chain.
// A operands register-pipelined at distance 2. Last 3 K-steps peeled.
// ---------------------------------------------------------------------------
__global__ __launch_bounds__(128) void step_kernel(const unsigned short* __restrict__ hA,
                                                   const unsigned short* __restrict__ xA,
                                                   const unsigned short* __restrict__ Wp,
                                                   const float* __restrict__ bvec,
                                                   unsigned short* __restrict__ hAnext,
                                                   float* __restrict__ accum) {
  __shared__ float ls[4 * 16 * 64];            // 16 KB: per-wave 16x64 result
  __shared__ unsigned short bsh[8 * 2048];     // 32 KB: 8-deep B chunk ring
  const int tid = threadIdx.x;
  const int wave = tid >> 5, lane = tid & 31;
  const int mt = blockIdx.y * 4 + wave;        // global 16-row tile (0..31)
  const int ntb = blockIdx.x * 4;              // base 16-col tile

  // B staging: thread covers 32 bytes of the 4 KB chunk (tile sti, 2 x b128)
  const int sti = tid >> 5;                    // tile 0..3
  const int spos = (tid & 31) << 4;            // element offset within tile
  const unsigned short* gstage =
      Wp + (((size_t)((ntb + sti) * KT_ALL)) << 9) + spos;

  v8f acc[4];
#pragma unroll
  for (int i = 0; i < 4; ++i) acc[i] = (v8f){0.f,0.f,0.f,0.f,0.f,0.f,0.f,0.f};

  auto loadA = [&](int kt) -> v16bf {
    const unsigned short* ap =
        (kt < KT_H) ? hA + (((size_t)(mt * KT_H + kt)) << 9) + lane * 16
                    : xA + (((size_t)(mt * KT_X + (kt - KT_H))) << 9) + lane * 16;
    return load16bf(ap);
  };
  auto wmma4 = [&](v16bf a, v16bf b0, v16bf b1, v16bf b2, v16bf b3) {
    acc[0] = __builtin_amdgcn_wmma_f32_16x16x32_bf16(false, a, false, b0,
                                                     (short)0, acc[0], false, false);
    acc[1] = __builtin_amdgcn_wmma_f32_16x16x32_bf16(false, a, false, b1,
                                                     (short)0, acc[1], false, false);
    acc[2] = __builtin_amdgcn_wmma_f32_16x16x32_bf16(false, a, false, b2,
                                                     (short)0, acc[2], false, false);
    acc[3] = __builtin_amdgcn_wmma_f32_16x16x32_bf16(false, a, false, b3,
                                                     (short)0, acc[3], false, false);
  };

#if USE_ASYNC_LDS
  auto issueB = [&](int kt, int bufi) {
    const unsigned short* g = gstage + (size_t)kt * 512;
    unsigned short* l = bsh + bufi * 2048 + sti * 512 + spos;
    GASYNC_B128(g, l);
    GASYNC_B128(g + 8, l + 8);
  };
  issueB(0, 0);
  issueB(1, 1);
  issueB(2, 2);
  v16bf a0 = loadA(0);
  v16bf a1 = loadA(1);
  __builtin_amdgcn_s_wait_asynccnt(4);         // chunk 0 landed
  __syncthreads();
  v16bf bc0, bc1, bc2, bc3;
  {
    const unsigned short* base = bsh + lane * 16;    // chunk 0
    bc0 = load16bf_lds(base);
    bc1 = load16bf_lds(base + 512);
    bc2 = load16bf_lds(base + 1024);
    bc3 = load16bf_lds(base + 1536);
  }
#pragma unroll 4
  for (int kt = 0; kt < KT_ALL - 3; ++kt) {
    issueB(kt + 3, (kt + 3) & 7);
    v16bf a2 = loadA(kt + 2);                  // distance-2 A prefetch
    __builtin_amdgcn_s_wait_asynccnt(4);       // chunks <= kt+1 landed
    __syncthreads();                           // chunk kt+1 visible to all
    // ds-load chunk kt+1 fragments (consumed next iteration)
    const unsigned short* base = bsh + ((kt + 1) & 7) * 2048 + lane * 16;
    v16bf bn0 = load16bf_lds(base);
    v16bf bn1 = load16bf_lds(base + 512);
    v16bf bn2 = load16bf_lds(base + 1024);
    v16bf bn3 = load16bf_lds(base + 1536);
    wmma4(a0, bc0, bc1, bc2, bc3);             // chunk kt: already in regs
    bc0 = bn0; bc1 = bn1; bc2 = bn2; bc3 = bn3;
    a0 = a1; a1 = a2;
  }
  {  // kt = KT_ALL-3
    v16bf a2 = loadA(KT_ALL - 1);
    __builtin_amdgcn_s_wait_asynccnt(2);       // chunk KT_ALL-2 landed
    __syncthreads();
    const unsigned short* base = bsh + ((KT_ALL - 2) & 7) * 2048 + lane * 16;
    v16bf bn0 = load16bf_lds(base);
    v16bf bn1 = load16bf_lds(base + 512);
    v16bf bn2 = load16bf_lds(base + 1024);
    v16bf bn3 = load16bf_lds(base + 1536);
    wmma4(a0, bc0, bc1, bc2, bc3);
    bc0 = bn0; bc1 = bn1; bc2 = bn2; bc3 = bn3;
    a0 = a1; a1 = a2;
  }
  {  // kt = KT_ALL-2
    __builtin_amdgcn_s_wait_asynccnt(0);       // last chunk landed
    __syncthreads();
    const unsigned short* base = bsh + ((KT_ALL - 1) & 7) * 2048 + lane * 16;
    v16bf bn0 = load16bf_lds(base);
    v16bf bn1 = load16bf_lds(base + 512);
    v16bf bn2 = load16bf_lds(base + 1024);
    v16bf bn3 = load16bf_lds(base + 1536);
    wmma4(a0, bc0, bc1, bc2, bc3);
    bc0 = bn0; bc1 = bn1; bc2 = bn2; bc3 = bn3;
    a0 = a1;
  }
  // kt = KT_ALL-1
  wmma4(a0, bc0, bc1, bc2, bc3);
  __syncthreads();
#else
  for (int kt = 0; kt < KT_ALL; ++kt) {
    const unsigned short* g = gstage + (size_t)kt * 512;
    uint4 v0 = ((const uint4*)g)[0];
    uint4 v1 = ((const uint4*)(g + 8))[0];
    v16bf a_cur = loadA(kt);
    __syncthreads();
    unsigned short* l = bsh + sti * 512 + spos;
    ((uint4*)l)[0] = v0;
    ((uint4*)(l + 8))[0] = v1;
    __syncthreads();
    const unsigned short* base = bsh + lane * 16;
    wmma4(a_cur, load16bf_lds(base), load16bf_lds(base + 512),
          load16bf_lds(base + 1024), load16bf_lds(base + 1536));
  }
  __syncthreads();
#endif

  // bias + relu -> LDS (C layout: row = v + 8*(lane>=16), col = lane&15)
  int rlo = lane & 15, hi = lane >> 4;
#pragma unroll
  for (int i = 0; i < 4; ++i) {
    int gn = blockIdx.x * 64 + i * 16 + rlo;
    float bias = bvec[gn];
#pragma unroll
    for (int v = 0; v < 8; ++v) {
      float val = fmaxf(acc[i][v] + bias, 0.f);
      ls[wave * 1024 + (v + (hi << 3)) * 64 + i * 16 + rlo] = val;
    }
  }
  __syncthreads();

  // activity loss: sum h_next columns >= SSIZE
  float part = 0.f;
  for (int e = tid; e < 4096; e += 128) {
    int gcol = blockIdx.x * 64 + (e & 63);
    if (gcol >= SSIZE) part += ls[e];
  }
#pragma unroll
  for (int off = 16; off; off >>= 1) part += __shfl_xor(part, off, 32);
  if (lane == 0) atomicAdd(&accum[1], part);

  // repack relu output into next-step A layout
#pragma unroll
  for (int kt2 = 0; kt2 < 2; ++kt2) {
    PK16 p;
#pragma unroll
    for (int j = 0; j < 16; ++j) {
      int kl = kt2 * 32 + kpatA(lane, j);
      p.s[j] = f2bf(ls[wave * 1024 + rlo * 64 + kl]);
    }
    uint4* dst = (uint4*)(hAnext +
        (((size_t)(mt * KT_H + blockIdx.x * 2 + kt2)) << 9) + lane * 16);
    dst[0] = p.q[0]; dst[1] = p.q[1];
  }
}

// ---------------------------------------------------------------------------
// Fused logits GEMM + log-softmax + cross-entropy (register-pipelined,
// final iteration peeled). Grid 32, 256 threads.
// ---------------------------------------------------------------------------
__global__ __launch_bounds__(256) void logits_kernel(const unsigned short* __restrict__ hAn,
                                                     const unsigned short* __restrict__ WoutP,
                                                     const float* __restrict__ bout,
                                                     const float* __restrict__ y,
                                                     float* __restrict__ accum,
                                                     int t) {
  __shared__ float lsd[16 * 512];
  __shared__ float red[256];
  int wave = threadIdx.x >> 5, lane = threadIdx.x & 31;
  int mt = blockIdx.x;
  int ntb = wave * 4;

  v8f acc[4];
#pragma unroll
  for (int i = 0; i < 4; ++i) acc[i] = (v8f){0.f,0.f,0.f,0.f,0.f,0.f,0.f,0.f};

  const unsigned short* aBase = hAn + (((size_t)(mt * KT_H)) << 9) + lane * 16;

  v16bf a_cur = load16bf(aBase);
  v16bf b_cur[4];
#pragma unroll
  for (int i = 0; i < 4; ++i)
    b_cur[i] = load16bf(WoutP + (((size_t)((ntb + i) * KT_H)) << 9) + lane * 16);

  for (int kt = 0; kt < KT_H - 1; ++kt) {
    v16bf a_nxt = load16bf(aBase + (size_t)(kt + 1) * 512);
    v16bf b_nxt[4];
#pragma unroll
    for (int i = 0; i < 4; ++i)
      b_nxt[i] = load16bf(WoutP + (((size_t)((ntb + i) * KT_H + kt + 1)) << 9) + lane * 16);
#pragma unroll
    for (int i = 0; i < 4; ++i)
      acc[i] = __builtin_amdgcn_wmma_f32_16x16x32_bf16(false, a_cur, false, b_cur[i],
                                                       (short)0, acc[i], false, false);
    a_cur = a_nxt;
#pragma unroll
    for (int i = 0; i < 4; ++i) b_cur[i] = b_nxt[i];
  }
#pragma unroll
  for (int i = 0; i < 4; ++i)
    acc[i] = __builtin_amdgcn_wmma_f32_16x16x32_bf16(false, a_cur, false, b_cur[i],
                                                     (short)0, acc[i], false, false);

  int rlo = lane & 15, hi = lane >> 4;
#pragma unroll
  for (int i = 0; i < 4; ++i) {
    int gn = wave * 64 + i * 16 + rlo;
    float bias = bout[gn];
#pragma unroll
    for (int v = 0; v < 8; ++v)
      lsd[(v + (hi << 3)) * 512 + gn] = acc[i][v] + bias;
  }
  __syncthreads();

  // log-softmax + CE: 16 threads per row
  int r = threadIdx.x >> 4, l16 = threadIdx.x & 15;
  float mx = -3.0e38f;
  for (int c = l16; c < 512; c += 16) mx = fmaxf(mx, lsd[r * 512 + c]);
  red[threadIdx.x] = mx; __syncthreads();
#pragma unroll
  for (int off = 8; off; off >>= 1) {
    if (l16 < off) red[threadIdx.x] = fmaxf(red[threadIdx.x], red[threadIdx.x + off]);
    __syncthreads();
  }
  mx = red[r << 4];
  __syncthreads();

  float se = 0.f;
  for (int c = l16; c < 512; c += 16) se += __expf(lsd[r * 512 + c] - mx);
  red[threadIdx.x] = se; __syncthreads();
#pragma unroll
  for (int off = 8; off; off >>= 1) {
    if (l16 < off) red[threadIdx.x] += red[threadIdx.x + off];
    __syncthreads();
  }
  float lse = __logf(red[r << 4]);
  __syncthreads();

  const float* yr = y + (((size_t)(mt * 16 + r)) * TSTEPS + t) * SSIZE;
  float ce = 0.f;
  for (int c = l16; c < 512; c += 16) ce += yr[c] * (lsd[r * 512 + c] - mx - lse);
  red[threadIdx.x] = ce; __syncthreads();
#pragma unroll
  for (int off = 8; off; off >>= 1) {
    if (l16 < off) red[threadIdx.x] += red[threadIdx.x + off];
    __syncthreads();
  }
  if (l16 == 0 && t >= 5) atomicAdd(&accum[0], -red[threadIdx.x]);
}

// ---------------------------------------------------------------------------
// Combine accumulators into the scalar loss
// ---------------------------------------------------------------------------
__global__ void final_kernel(const float* __restrict__ accum, float* __restrict__ out) {
  float xe = accum[0] / (123.f * 512.f);
  float wl = 0.1f * (accum[2] / (1536.f * 1024.f) +
                     accum[3] / (1024.f * 1536.f) +
                     accum[4] / (1024.f * 1024.f));
  float al = 0.1f * accum[1] / (128.f * 512.f * 1024.f);
  out[0] = xe + wl + al;
}

// ---------------------------------------------------------------------------
// Host launcher
// ---------------------------------------------------------------------------
extern "C" void kernel_launch(void* const* d_in, const int* in_sizes, int n_in,
                              void* d_out, int out_size, void* d_ws, size_t ws_size,
                              hipStream_t stream) {
  const float* x     = (const float*)d_in[0];
  const float* y     = (const float*)d_in[1];
  const float* Wh    = (const float*)d_in[2];
  const float* Win   = (const float*)d_in[3];
  const float* b     = (const float*)d_in[4];
  const float* Wout  = (const float*)d_in[5];
  const float* bout  = (const float*)d_in[6];

  float* accum = (float*)d_ws;
  unsigned short* Wp    = (unsigned short*)((char*)d_ws + 256);
  unsigned short* WoutP = Wp + (size_t)KTOT * HSIZE;
  unsigned short* hA0   = WoutP + (size_t)HSIZE * SSIZE;
  unsigned short* hA1   = hA0 + (size_t)BATCH * HSIZE;
  unsigned short* xA    = hA1 + (size_t)BATCH * HSIZE;

  zero_kernel<<<512, 256, 0, stream>>>((unsigned int*)hA0,
                                       (int)((size_t)BATCH * HSIZE / 2), accum);
  packW_kernel<<<768, 256, 0, stream>>>(Wh, Win, Wp);
  packWout_kernel<<<192, 256, 0, stream>>>(Wout, WoutP);
  wloss_kernel<<<512, 256, 0, stream>>>(Wh, accum);

  for (int t = 0; t < TSTEPS; ++t) {
    const unsigned short* hin = (t & 1) ? hA1 : hA0;
    unsigned short* hout      = (t & 1) ? hA0 : hA1;
    packx_kernel<<<64, 256, 0, stream>>>(x, xA, t);
    step_kernel<<<dim3(24, 8), 128, 0, stream>>>(hin, xA, Wp, b, hout, accum);
    logits_kernel<<<32, 256, 0, stream>>>(hout, WoutP, bout, y, accum, t);
  }
  final_kernel<<<1, 1, 0, stream>>>(accum, (float*)d_out);
}
